// QloraLinear_14121852469691
// MI455X (gfx1250) — compile-verified, matrix operand order checked by reading, not run on hardware
//
#include <hip/hip_runtime.h>
#include <hip/hip_bf16.h>
#include <stdint.h>

// ---------------------------------------------------------------------------
// QLoRA linear for MI455X (gfx1250, wave32):
//   out = x @ (dequant_nf4(W) + 2*B@A)^T
// LoRA folded into dequantized bf16 weight; bf16 WMMA GEMM with async
// global->LDS staging (ASYNCcnt) and double buffering.
// Workspace: [0,32MB) W_eff bf16 [4096x4096]; [32MB,96MB) x bf16.
// ---------------------------------------------------------------------------

#define IN_F   4096
#define OUT_F  4096
#define MROWS  8192          // BATCH*SEQ
#define KSTEP  32            // K per WMMA / LDS stage
#define ROWB   80            // LDS bytes/row: 64 data + 16 pad (bank-conflict-free)
#define NKIT   (IN_F / KSTEP)
#define MBLK   128           // block tile M
#define NBLK   256           // block tile N (8 waves: 2x4 of 64x64)
#define BUFB   ((MBLK + NBLK) * ROWB)   // 30720 B per buffer

typedef __bf16 bf16_t;
typedef __attribute__((ext_vector_type(16))) __bf16 v16bf;
typedef __attribute__((ext_vector_type(8)))  float  v8f;

__constant__ float NF4C[16] = {
    -1.0f, -0.6961928009986877f, -0.5250730514526367f, -0.39491748809814453f,
    -0.28444138169288635f, -0.18477343022823334f, -0.09105003625154495f, 0.0f,
    0.07958029955625534f, 0.16093020141124725f, 0.24611230194568634f, 0.33791524171829224f,
    0.44070982933044434f, 0.5626170039176941f, 0.7229568362236023f, 1.0f};

__device__ __forceinline__ uint32_t f2bf(float f) {
    uint32_t u = __builtin_bit_cast(uint32_t, f);
    uint32_t r = u + 0x7FFFu + ((u >> 16) & 1u);   // round-to-nearest-even
    return r >> 16;
}

// ---------------------------------------------------------------------------
// Kernel 1: NF4 double-dequant + LoRA fold -> bf16 W_eff [OUT_F, IN_F]
// ---------------------------------------------------------------------------
__global__ __launch_bounds__(256) void dequant_lora_fold(
    const int* __restrict__ qd, const int* __restrict__ qs,
    const float* __restrict__ qf, const float* __restrict__ meanp,
    const float* __restrict__ lA, const float* __restrict__ lB,
    uint16_t* __restrict__ weff) {
    __shared__ float lut[16];
    const int t = threadIdx.x;
    const int o = blockIdx.x;
    if (t < 16) lut[t] = NF4C[t];
    __syncthreads();

    const int k0 = t * 16;
    const int si = o * 64 + (k0 >> 6);
    const float s = (float)qs[si] / qf[si >> 8] + meanp[0];

    const int4 p0 = *(const int4*)(qd + (size_t)o * 2048 + t * 8);
    const int4 p1 = *(const int4*)(qd + (size_t)o * 2048 + t * 8 + 4);
    int pk[8] = {p0.x, p0.y, p0.z, p0.w, p1.x, p1.y, p1.z, p1.w};

    float l[16];
#pragma unroll
    for (int i = 0; i < 16; ++i) l[i] = 0.0f;
#pragma unroll
    for (int r = 0; r < 16; ++r) {
        const float br = lB[o * 16 + r];
        const float* ap = lA + (size_t)r * IN_F + k0;
#pragma unroll
        for (int i = 0; i < 16; ++i) l[i] = fmaf(br, ap[i], l[i]);
    }

    uint32_t outw[8];
#pragma unroll
    for (int j = 0; j < 8; ++j) {
        const int b = pk[j] & 0xFF;
        const float e0 = lut[b >> 4] * s + 2.0f * l[2 * j];      // SCALING = 32/16
        const float e1 = lut[b & 15] * s + 2.0f * l[2 * j + 1];
        outw[j] = f2bf(e0) | (f2bf(e1) << 16);
    }
    uint4* dst = (uint4*)(weff + (size_t)o * IN_F + k0);
    uint4 a; a.x = outw[0]; a.y = outw[1]; a.z = outw[2]; a.w = outw[3];
    uint4 b; b.x = outw[4]; b.y = outw[5]; b.z = outw[6]; b.w = outw[7];
    dst[0] = a; dst[1] = b;
}

// ---------------------------------------------------------------------------
// Kernel 2: x f32 -> bf16
// ---------------------------------------------------------------------------
__global__ __launch_bounds__(256) void x_to_bf16(const float* __restrict__ x,
                                                 uint16_t* __restrict__ xb) {
    const size_t i = ((size_t)blockIdx.x * 256 + threadIdx.x) * 8;
    const float4 a = *(const float4*)(x + i);
    const float4 b = *(const float4*)(x + i + 4);
    uint4 w;
    w.x = f2bf(a.x) | (f2bf(a.y) << 16);
    w.y = f2bf(a.z) | (f2bf(a.w) << 16);
    w.z = f2bf(b.x) | (f2bf(b.y) << 16);
    w.w = f2bf(b.z) | (f2bf(b.w) << 16);
    *(uint4*)(xb + i) = w;
}

// ---------------------------------------------------------------------------
// Kernel 3: bf16 WMMA GEMM. Block tile 128(M)x256(N), K-step 32, 8 waves,
// each wave 64x64 via 4x4 v_wmma_f32_16x16x32_bf16. Async LDS double buffer.
// ---------------------------------------------------------------------------
__device__ __forceinline__ void async_load_b128(uint32_t lds_off, const void* gaddr) {
    asm volatile("global_load_async_to_lds_b128 %0, %1, off"
                 :: "v"(lds_off), "v"(gaddr) : "memory");
}
__device__ __forceinline__ void wait_asynccnt0() {
    asm volatile("s_wait_asynccnt 0x0" ::: "memory");
}

union Frag {
    v16bf v;
    uint4 q[2];
};

__global__ __launch_bounds__(256) void qlora_gemm(const uint16_t* __restrict__ xb,
                                                  const uint16_t* __restrict__ wb,
                                                  float* __restrict__ out) {
    // per buffer: x tile 128 rows then w tile 256 rows, ROWB bytes each -> 60KB total
    __shared__ uint8_t smem[2][BUFB];

    const int tid    = threadIdx.x;
    const int wid    = tid >> 5;
    const int lane   = tid & 31;
    const int nBlock = blockIdx.x * NBLK;
    const int mBlock = blockIdx.y * MBLK;
    const int mW = (wid >> 2) * 64;   // wave tile 64x64
    const int nW = (wid & 3) * 64;
    const int rA  = lane & 15;
    const int sel = lane & 16;        // byte select for lane-high K half

    uint32_t ldsCur = (uint32_t)(uintptr_t)&smem[0][0];
    uint32_t ldsNxt = (uint32_t)(uintptr_t)&smem[1][0];
    const uint8_t* pCur = &smem[0][0];
    const uint8_t* pNxt = &smem[1][0];

    v8f acc[4][4];
#pragma unroll
    for (int mt = 0; mt < 4; ++mt)
#pragma unroll
        for (int nt = 0; nt < 4; ++nt)
#pragma unroll
            for (int e = 0; e < 8; ++e) acc[mt][nt][e] = 0.0f;

    // stage one K-step: x 512 chunks + w 1024 chunks of 16B -> 6 per thread
    auto stage = [&](uint32_t ldsBuf, int ks) {
#pragma unroll
        for (int i = 0; i < 2; ++i) {           // x tile: 128 rows x 64B
            const int ch = tid + i * 256;
            const int r  = ch >> 2;
            const int cb = (ch & 3) * 16;
            const char* gx = (const char*)(xb + (size_t)(mBlock + r) * IN_F + ks) + cb;
            async_load_b128(ldsBuf + r * ROWB + cb, gx);
        }
#pragma unroll
        for (int i = 0; i < 4; ++i) {           // w tile: 256 rows x 64B
            const int ch = tid + i * 256;
            const int r  = ch >> 2;
            const int cb = (ch & 3) * 16;
            const char* gw = (const char*)(wb + (size_t)(nBlock + r) * IN_F + ks) + cb;
            async_load_b128(ldsBuf + MBLK * ROWB + r * ROWB + cb, gw);
        }
    };

    stage(ldsCur, 0);
    wait_asynccnt0();
    __syncthreads();

#pragma unroll 1
    for (int ks = 0; ks < NKIT; ++ks) {
        if (ks + 1 < NKIT) stage(ldsNxt, (ks + 1) * KSTEP);

        // A fragments: lane<16 holds K{0..7,16..23}, lane>=16 K{8..15,24..31}
        Frag a[4];
#pragma unroll
        for (int mt = 0; mt < 4; ++mt) {
            const uint8_t* p = pCur + (mW + mt * 16 + rA) * ROWB + sel;
            a[mt].q[0] = *(const uint4*)(p);
            a[mt].q[1] = *(const uint4*)(p + 32);
        }
        // B fragments double-buffered: next group's ds_loads overlap this
        // group's 4-WMMA chain (no full dscnt drain between groups)
        Frag b[2];
        {
            const uint8_t* p = pCur + MBLK * ROWB + (nW + rA) * ROWB + sel;
            b[0].q[0] = *(const uint4*)(p);
            b[0].q[1] = *(const uint4*)(p + 32);
        }
#pragma unroll
        for (int nt = 0; nt < 4; ++nt) {
            if (nt + 1 < 4) {
                const uint8_t* p =
                    pCur + MBLK * ROWB + (nW + (nt + 1) * 16 + rA) * ROWB + sel;
                b[(nt + 1) & 1].q[0] = *(const uint4*)(p);
                b[(nt + 1) & 1].q[1] = *(const uint4*)(p + 32);
            }
#pragma unroll
            for (int mt = 0; mt < 4; ++mt)
                acc[mt][nt] = __builtin_amdgcn_wmma_f32_16x16x32_bf16(
                    false, a[mt].v, false, b[nt & 1].v,
                    (short)0, acc[mt][nt], false, false);
        }

        wait_asynccnt0();
        __syncthreads();

        const uint32_t tl = ldsCur; ldsCur = ldsNxt; ldsNxt = tl;
        const uint8_t* tp = pCur; pCur = pNxt; pNxt = tp;
    }

    // D layout: lanes 0-15 N=lane, M=vgpr; lanes 16-31 N=lane-16, M=8+vgpr
#pragma unroll
    for (int mt = 0; mt < 4; ++mt)
#pragma unroll
        for (int nt = 0; nt < 4; ++nt) {
            const int n  = nBlock + nW + nt * 16 + rA;
            const int m0 = mBlock + mW + mt * 16 + ((lane & 16) ? 8 : 0);
#pragma unroll
            for (int v = 0; v < 8; ++v)
                out[(size_t)(m0 + v) * OUT_F + n] = acc[mt][nt][v];
        }
}

// ---------------------------------------------------------------------------
extern "C" void kernel_launch(void* const* d_in, const int* in_sizes, int n_in,
                              void* d_out, int out_size, void* d_ws, size_t ws_size,
                              hipStream_t stream) {
    const float* x    = (const float*)d_in[0];
    const int*   qd   = (const int*)d_in[1];
    const int*   qs   = (const int*)d_in[2];
    const float* qf   = (const float*)d_in[3];
    const float* mean = (const float*)d_in[4];
    const float* lA   = (const float*)d_in[5];
    const float* lB   = (const float*)d_in[6];
    float* out = (float*)d_out;

    uint16_t* weff = (uint16_t*)d_ws;                                      // 32 MB
    uint16_t* xbf  = (uint16_t*)((char*)d_ws + (size_t)OUT_F * IN_F * 2);  // 64 MB

    dequant_lora_fold<<<OUT_F, 256, 0, stream>>>(qd, qs, qf, mean, lA, lB, weff);
    x_to_bf16<<<((size_t)MROWS * IN_F) / (256 * 8), 256, 0, stream>>>(x, xbf);

    dim3 grid(OUT_F / NBLK, MROWS / MBLK);
    qlora_gemm<<<grid, 256, 0, stream>>>(xbf, weff, out);
}
